// Unidir_atten_38276748542489
// MI455X (gfx1250) — matrix operand order repaired and checked by reading, not compiled, not run on hardware
//
#include <hip/hip_runtime.h>
#include <hip/hip_bf16.h>

// ---------------------------------------------------------------------------
// SeqAttnMatch (relu_FC proj, softmax over L2, concat_dot_diff) for MI455X.
// bf16 WMMA (v_wmma_f32_16x16x32_bf16) with f32 accumulation; scores stripe
// held in LDS (flash-style); Xp stripe staged via global_load_async_to_lds;
// f32->bf16 via packed v_cvt_pk_bf16_f32.
// B=16, L1=L2=1024, H=768.
// ---------------------------------------------------------------------------

#define Bv   16
#define L1v  1024
#define L2v  1024
#define Hv   768

typedef __attribute__((ext_vector_type(16))) __bf16        v16bf;
typedef __attribute__((ext_vector_type(8)))  float         v8f;
typedef __attribute__((ext_vector_type(4)))  unsigned int  v4u;
typedef __attribute__((ext_vector_type(8)))  unsigned int  v8u;

// Packed f32->bf16 (RNE) : dst[15:0]=cvt(lo), dst[31:16]=cvt(hi)
__device__ __forceinline__ unsigned int cvt_pk_bf16(float lo, float hi) {
  unsigned int r;
  asm("v_cvt_pk_bf16_f32 %0, %1, %2" : "=v"(r) : "v"(lo), "v"(hi));
  return r;
}
__device__ __forceinline__ unsigned short f32_to_bf16(float f) {
  return (unsigned short)cvt_pk_bf16(f, 0.0f);
}

// Async copy global -> LDS (16B per lane), tracked on ASYNCcnt.
__device__ __forceinline__ void async_ld_b128(unsigned int lds_off,
                                              const void* gptr) {
  asm volatile("global_load_async_to_lds_b128 %0, %1, off"
               :: "v"(lds_off), "v"((unsigned long long)(uintptr_t)gptr)
               : "memory");
}
__device__ __forceinline__ void wait_async0() {
  asm volatile("s_wait_asynccnt 0" ::: "memory");
}

__device__ __forceinline__ v8f wmma_bf16(v16bf a, v16bf b, v8f c) {
  return __builtin_amdgcn_wmma_f32_16x16x32_bf16(
      /*neg_a=*/false, a, /*neg_b=*/false, b,
      /*c_mod=*/(short)0, c, /*reuse_a=*/false, /*reuse_b=*/false);
}

// A fragment (16x32 MxK, bf16 row-major, stride lda elements).
// lane L: row = L&15, kbase = 8*(L>>4); VGPR0-3 = K[kbase..kbase+7],
// VGPR4-7 = K[16+kbase..16+kbase+7]  -> two 16B contiguous loads.
__device__ __forceinline__ v16bf load_a_bf16(const unsigned short* base, int lda) {
  int lane = threadIdx.x & 31;
  const unsigned short* p = base + (size_t)(lane & 15) * lda + ((lane >> 4) * 8);
  v4u lo = *(const v4u*)(p);
  v4u hi = *(const v4u*)(p + 16);
  v8u v = {lo[0], lo[1], lo[2], lo[3], hi[0], hi[1], hi[2], hi[3]};
  return __builtin_bit_cast(v16bf, v);
}

// Same A fragment, but source is f32 row-major (packed convert on the fly).
__device__ __forceinline__ v16bf load_a_f32(const float* base, int lda) {
  int lane = threadIdx.x & 31;
  const float* p = base + (size_t)(lane & 15) * lda + ((lane >> 4) * 8);
  unsigned int o[8];
#pragma unroll
  for (int h = 0; h < 2; ++h) {
    const float* q = p + h * 16;
#pragma unroll
    for (int e = 0; e < 4; ++e)
      o[h * 4 + e] = cvt_pk_bf16(q[2 * e], q[2 * e + 1]);
  }
  v8u v = {o[0], o[1], o[2], o[3], o[4], o[5], o[6], o[7]};
  return __builtin_bit_cast(v16bf, v);
}

// B fragment (32x16 KxN) loaded from the *transposed* operand BT[N x K]
// row-major, stride ldk. lane L: n = L&15, kb = 8*(L>>4); K[kb..kb+7], K[16+kb..16+kb+7].
__device__ __forceinline__ v16bf load_b_bf16(const unsigned short* baseT, int ldk) {
  int lane = threadIdx.x & 31;
  const unsigned short* p = baseT + (size_t)(lane & 15) * ldk + ((lane >> 4) * 8);
  v4u lo = *(const v4u*)(p);
  v4u hi = *(const v4u*)(p + 16);
  v8u v = {lo[0], lo[1], lo[2], lo[3], hi[0], hi[1], hi[2], hi[3]};
  return __builtin_bit_cast(v16bf, v);
}

// ---------------------------------------------------------------------------
// Kernel 1: W (f32 [H,H]) -> bf16 (paired stores)
// ---------------------------------------------------------------------------
__global__ __launch_bounds__(256) void cvt_w_kernel(const float* __restrict__ W,
                                                    unsigned int* __restrict__ Wbf) {
  int i = blockIdx.x * 256 + threadIdx.x;      // handles 4 f32 -> 2 u32
  const float* s = W + (size_t)i * 4;
  Wbf[i * 2]     = cvt_pk_bf16(s[0], s[1]);
  Wbf[i * 2 + 1] = cvt_pk_bf16(s[2], s[3]);
}

// ---------------------------------------------------------------------------
// Kernel 2: x2T[b][d][j] = bf16(x2[b][j][d])  (per-batch transpose, LDS tiled)
// grid = B * (H/32) * (L2/32), block 256 (32x8)
// ---------------------------------------------------------------------------
__global__ __launch_bounds__(256) void transpose_cvt_kernel(const float* __restrict__ x2,
                                                            unsigned short* __restrict__ x2T) {
  __shared__ float tile[32][33];
  int bid = blockIdx.x;
  int jt = bid & 31;            // L2/32 = 32
  bid >>= 5;
  int dt = bid % (Hv / 32);
  int b  = bid / (Hv / 32);
  int tx = threadIdx.x & 31, ty = threadIdx.x >> 5;
  const float* src = x2 + ((size_t)b * L2v + jt * 32) * Hv + dt * 32;
#pragma unroll
  for (int s = 0; s < 4; ++s) {
    int j = ty + s * 8;
    tile[j][tx] = src[(size_t)j * Hv + tx];
  }
  __syncthreads();
  // write 32-bit pairs along j (contiguous in x2T)
  unsigned int* dst = (unsigned int*)(x2T + ((size_t)b * Hv + dt * 32) * L2v + jt * 32);
  int txp = tx & 15;            // 16 pair-columns
  int drow = (ty * 4) + (tx >> 4) * 2;   // covers d = 0..31 over two steps
#pragma unroll
  for (int s = 0; s < 2; ++s) {
    int d = drow + s;
    dst[(size_t)d * (L2v / 2) + txp] = cvt_pk_bf16(tile[2 * txp][d], tile[2 * txp + 1][d]);
  }
}

// ---------------------------------------------------------------------------
// Kernel 3: Out[m,h] = bf16(relu( sum_d X[m,d]*W[h,d] + bias[h] ))
// X: f32 [16384 x 768]; Wbf row-major already IS the transposed B operand.
// grid = (16384/128, 768/64), block 256 (8 waves, 4(M) x 2(N), 32x32 each)
// ---------------------------------------------------------------------------
__global__ __launch_bounds__(256) void proj_relu_kernel(const float* __restrict__ X,
                                                        const unsigned short* __restrict__ Wbf,
                                                        const float* __restrict__ bias,
                                                        unsigned short* __restrict__ Out) {
  int w = threadIdx.x >> 5;
  int lane = threadIdx.x & 31;
  int m0 = blockIdx.x * 128 + (w & 3) * 32;
  int n0 = blockIdx.y * 64 + (w >> 2) * 32;

  v8f acc[2][2] = {};
  for (int kk = 0; kk < Hv; kk += 32) {
    v16bf a0 = load_a_f32(X + (size_t)m0 * Hv + kk, Hv);
    v16bf a1 = load_a_f32(X + (size_t)(m0 + 16) * Hv + kk, Hv);
    v16bf b0 = load_b_bf16(Wbf + (size_t)n0 * Hv + kk, Hv);
    v16bf b1 = load_b_bf16(Wbf + (size_t)(n0 + 16) * Hv + kk, Hv);
    acc[0][0] = wmma_bf16(a0, b0, acc[0][0]);
    acc[0][1] = wmma_bf16(a0, b1, acc[0][1]);
    acc[1][0] = wmma_bf16(a1, b0, acc[1][0]);
    acc[1][1] = wmma_bf16(a1, b1, acc[1][1]);
  }
  int col = lane & 15, rbase = (lane >> 4) * 8;
#pragma unroll
  for (int mt = 0; mt < 2; ++mt)
#pragma unroll
    for (int nt = 0; nt < 2; ++nt) {
      int n = n0 + nt * 16 + col;
      float bv = bias[n];
#pragma unroll
      for (int r = 0; r < 8; ++r) {
        int m = m0 + mt * 16 + rbase + r;
        float v = acc[mt][nt][r] + bv;
        v = v > 0.0f ? v : 0.0f;
        Out[(size_t)m * Hv + n] = f32_to_bf16(v);
      }
    }
}

// ---------------------------------------------------------------------------
// Kernel 4: fused attention + concat epilogue.
// One WG per (batch, 32-row stripe of L1).
// LDS: 32x1024 f32 scores (128K) | 32x1024 bf16 alpha (64K) |
//      32x768 bf16 Xp stripe (48K, async-staged) | reduction scratch.
// ---------------------------------------------------------------------------
__global__ __launch_bounds__(256) void attn_fused_kernel(
    const unsigned short* __restrict__ Xp, const unsigned short* __restrict__ Yp,
    const unsigned short* __restrict__ x2T, const float* __restrict__ x1,
    const int* __restrict__ x2_mask, float* __restrict__ out) {
  extern __shared__ char smem[];
  float*          S    = (float*)smem;                               // 131072 B
  unsigned short* Alp  = (unsigned short*)(smem + 131072);           //  65536 B
  unsigned short* Xlds = (unsigned short*)(smem + 131072 + 65536);   //  49152 B
  float*          red  = (float*)(smem + 131072 + 65536 + 49152);    //   1024 B
  float*          rowv = red + 256;                                  //    128 B

  int b  = blockIdx.x >> 5;        // L1/32 = 32 stripes
  int i0 = (blockIdx.x & 31) * 32;
  int tid = threadIdx.x;
  int w = tid >> 5, lane = tid & 31;
  int col = lane & 15, rbase = (lane >> 4) * 8;

  const unsigned short* Xpb = Xp + ((size_t)b * L1v + i0) * Hv;
  const unsigned short* Ypb = Yp + (size_t)b * L2v * Hv;

  // ---- Stage Xp stripe (32x768 bf16 = 48KB contiguous) into LDS -------
  {
    const char* gsrc = (const char*)Xpb;
    char* ldst = (char*)Xlds;
#pragma unroll
    for (int t = 0; t < 12; ++t) {         // 12 * 256 * 16B = 49152B
      int off = (t * 256 + tid) * 16;
      async_ld_b128((unsigned int)(uintptr_t)(ldst + off), gsrc + off);
    }
    wait_async0();
  }
  __syncthreads();

  // ---- Phase 1: scores stripe into LDS --------------------------------
  for (int q = 0; q < 8; ++q) {
    int j0 = (w + q * 8) * 16;
#pragma unroll
    for (int mt = 0; mt < 2; ++mt) {
      v8f acc = {};
      for (int kk = 0; kk < Hv; kk += 32) {
        v16bf a  = load_a_bf16(Xlds + (size_t)(mt * 16) * Hv + kk, Hv);
        v16bf bb = load_b_bf16(Ypb + (size_t)j0 * Hv + kk, Hv);
        acc = wmma_bf16(a, bb, acc);
      }
#pragma unroll
      for (int r = 0; r < 8; ++r)
        S[(mt * 16 + rbase + r) * 1024 + (j0 + col)] = acc[r];
    }
  }
  __syncthreads();

  // ---- Phase 2: masked softmax (256 thr = 32 rows x 8 segments) -------
  {
    int row = tid >> 3, seg = tid & 7;
    const int* mrow = x2_mask + (size_t)b * L2v + seg * 128;
    float* srow = S + row * 1024 + seg * 128;
    float mx = -__builtin_inff();
    for (int j = 0; j < 128; ++j) {
      float s = mrow[j] ? -__builtin_inff() : srow[j];
      srow[j] = s;
      mx = fmaxf(mx, s);
    }
    red[row * 8 + seg] = mx;
    __syncthreads();
    if (seg == 0) {
      float m2 = red[row * 8];
#pragma unroll
      for (int t = 1; t < 8; ++t) m2 = fmaxf(m2, red[row * 8 + t]);
      rowv[row] = m2;
    }
    __syncthreads();
    float rmax = rowv[row];
    float sum = 0.0f;
    for (int j = 0; j < 128; ++j) {
      float p = __expf(srow[j] - rmax);
      srow[j] = p;
      sum += p;
    }
    __syncthreads();
    red[row * 8 + seg] = sum;
    __syncthreads();
    if (seg == 0) {
      float s2 = red[row * 8];
#pragma unroll
      for (int t = 1; t < 8; ++t) s2 += red[row * 8 + t];
      rowv[row] = 1.0f / s2;
    }
    __syncthreads();
    float inv = rowv[row];
    unsigned int* arow = (unsigned int*)(Alp + row * 1024 + seg * 128);
    for (int j = 0; j < 64; ++j)
      arow[j] = cvt_pk_bf16(srow[2 * j] * inv, srow[2 * j + 1] * inv);
  }
  __syncthreads();

  // ---- Phase 3: att = alpha . x2 ; concat epilogue --------------------
  const unsigned short* x2Tb = x2T + (size_t)b * Hv * L2v;
  const float* x1b = x1 + ((size_t)b * L1v + i0) * Hv;
  float* outb = out + ((size_t)b * L1v + i0) * (4 * Hv);
  for (int q = 0; q < 6; ++q) {
    int d0 = (w + q * 8) * 16;    // 48 n-tiles over 8 waves
#pragma unroll
    for (int mt = 0; mt < 2; ++mt) {
      v8f acc = {};
      for (int kk = 0; kk < L2v; kk += 32) {
        v16bf a  = load_a_bf16(Alp + (size_t)(mt * 16) * 1024 + kk, 1024);
        v16bf bb = load_b_bf16(x2Tb + (size_t)d0 * L2v + kk, L2v);
        acc = wmma_bf16(a, bb, acc);
      }
      int d = d0 + col;
#pragma unroll
      for (int r = 0; r < 8; ++r) {
        int il = mt * 16 + rbase + r;
        float xv = x1b[(size_t)il * Hv + d];
        float at = acc[r];
        float* orow = outb + (size_t)il * (4 * Hv);
        orow[d]            = xv;
        orow[Hv + d]       = at;
        orow[2 * Hv + d]   = xv * at;
        orow[3 * Hv + d]   = xv - at;
      }
    }
  }
}

// ---------------------------------------------------------------------------
// Launch.  Inputs: 0=x1, 1=x2, 2=x1_mask(unused by output), 3=x2_mask,
// 4=W [H,H], 5=b [H].  Output: [B, L1, 4H] f32.
// Workspace (~73.2 MB): Wbf | Xp | Yp | x2T (all bf16).
// ---------------------------------------------------------------------------
extern "C" void kernel_launch(void* const* d_in, const int* in_sizes, int n_in,
                              void* d_out, int out_size, void* d_ws, size_t ws_size,
                              hipStream_t stream) {
  const float* x1      = (const float*)d_in[0];
  const float* x2      = (const float*)d_in[1];
  const int*   x2_mask = (const int*)d_in[3];
  const float* W       = (const float*)d_in[4];
  const float* bias    = (const float*)d_in[5];
  float* out = (float*)d_out;

  unsigned short* Wbf = (unsigned short*)d_ws;
  unsigned short* Xp  = (unsigned short*)((char*)d_ws + (size_t)Hv * Hv * 2);
  unsigned short* Yp  = Xp + (size_t)Bv * L1v * Hv;
  unsigned short* x2T = Yp + (size_t)Bv * L2v * Hv;

  cvt_w_kernel<<<(Hv * Hv) / 1024, 256, 0, stream>>>(W, (unsigned int*)Wbf);
  transpose_cvt_kernel<<<Bv * (Hv / 32) * (L2v / 32), 256, 0, stream>>>(x2, x2T);
  proj_relu_kernel<<<dim3((Bv * L1v) / 128, Hv / 64), 256, 0, stream>>>(x1, Wbf, bias, Xp);
  proj_relu_kernel<<<dim3((Bv * L2v) / 128, Hv / 64), 256, 0, stream>>>(x2, Wbf, bias, Yp);

  size_t smem = 131072 + 65536 + 49152 + 1024 + 128;  // 246,912 B (< 320K/WGP)
  attn_fused_kernel<<<Bv * (L1v / 32), 256, smem, stream>>>(Xp, Yp, x2T, x1, x2_mask, out);
}
